// AgeUGP_v1_18081812317001
// MI455X (gfx1250) — matrix operand order repaired and checked by reading, not compiled, not run on hardware
//
#include <hip/hip_runtime.h>
#include <hip/hip_bf16.h>

typedef __attribute__((ext_vector_type(16))) _Float16 v16h;
typedef __attribute__((ext_vector_type(8)))  _Float16 v8h;
typedef __attribute__((ext_vector_type(8)))  float    v8f;

#define B        128
#define N_SNPS   500000
#define N_NODES  600000
#define N_GENES  18000
#define DDIM     64
#define NF       8
#define KP       18048    // N_GENES padded to multiple of 64 (282 * 64)
#define BN_EPS   1e-5f

// ---------------- filt_mean = mean(filters, axis=0) ----------------
__global__ void k_filt_mean(const float* __restrict__ filters, float* __restrict__ fm) {
    int j = blockIdx.x * blockDim.x + threadIdx.x;
    if (j >= N_SNPS) return;
    float s = 0.f;
#pragma unroll
    for (int f = 0; f < NF; ++f) s += filters[(long)f * N_SNPS + j];
    fm[j] = s * (1.0f / NF);
}

// ---------------- gene boundaries via lower_bound (segment_ids sorted) ----------------
__global__ void k_gene_start(const int* __restrict__ seg, int* __restrict__ gs) {
    int g = blockIdx.x * blockDim.x + threadIdx.x;
    if (g > N_GENES) return;
    int lo = 0, hi = N_NODES;
    while (lo < hi) {
        int mid = (lo + hi) >> 1;
        if (seg[mid] < g) lo = mid + 1; else hi = mid;
    }
    gs[g] = lo;
}

// ---------------- gather + segment-sum -> sample_h (f16, B x KP, zero-padded) ----------------
// Thread order (b major) keeps each 2MB snp row L2-resident; ~38x cacheline reuse,
// so HBM traffic ~= one pass over snp (256 MB).
__global__ void k_gather(const float* __restrict__ snp, const int* __restrict__ ids,
                         const int* __restrict__ gs, const float* __restrict__ fm,
                         _Float16* __restrict__ Ah) {
    long t = (long)blockIdx.x * blockDim.x + threadIdx.x;
    if (t >= (long)B * KP) return;
    int b = (int)(t / KP);
    int g = (int)(t % KP);
    float s = 0.f;
    if (g < N_GENES) {
        int n0 = gs[g], n1 = gs[g + 1];
        const float* row = snp + (long)b * N_SNPS;
        for (int n = n0; n < n1; ++n) {
            int j = ids[n];
            s += row[j] * fm[j];
        }
    }
    Ah[(long)b * KP + g] = (_Float16)s;
}

// ---------------- f32 -> f16 convert with K padding ----------------
__global__ void k_cvt_f16(const float* __restrict__ src, _Float16* __restrict__ dst,
                          int rows, int K, int Kp) {
    long t = (long)blockIdx.x * blockDim.x + threadIdx.x;
    if (t >= (long)rows * Kp) return;
    int r = (int)(t / Kp);
    int k = (int)(t % Kp);
    dst[t] = (k < K) ? (_Float16)src[(long)r * K + k] : (_Float16)0.f;
}

// ---------------- WMMA GEMM (D = A * B^T) + bias + BN + ReLU ----------------
// A: M x Kp f16 row-major (M = 128 batch). Bw: N x Kp f16 row-major (N = 64 channels).
// One wave (32 threads) per 16x16 output tile. Layouts per CDNA5 ISA 7.12.2:
//   A lane: m = tile*16 + (lane&15); halves 0-7 = K[kb..kb+8), halves 8-15 = K[kb+16..kb+24), kb = (lane>>4)*8
//   B lane: n = tile*16 + (lane&15); halves 0-7 = K[kbB..kbB+8), halves 8-15 = K[kbB+16..kbB+24), kbB = (lane>>4)*8
//   D lane: n = lane&15; VGPR r holds m = r + 8*(lane>>4)
// Two independent accumulators (K interleave of 32) halve the serialized WMMA
// dependency chain — important because only 32 waves run this kernel.
template <typename OUT>
__global__ void k_gemm_bn_relu(const _Float16* __restrict__ Ah,
                               const _Float16* __restrict__ Bw,
                               const float* __restrict__ bias,
                               const float* __restrict__ gam,
                               const float* __restrict__ bet,
                               const float* __restrict__ mu,
                               const float* __restrict__ var,
                               OUT* __restrict__ out, int Kp) {
    int lane  = threadIdx.x;              // 0..31, one wave per block
    int Mtile = blockIdx.x;               // M/16 tiles
    int Ntile = blockIdx.y;               // N/16 tiles
    int mrow  = Mtile * 16 + (lane & 15);
    int nrow  = Ntile * 16 + (lane & 15);
    int kbA   = (lane >> 4) * 8;
    int kbB   = (lane >> 4) * 8;
    const _Float16* Abase = Ah + (long)mrow * Kp + kbA;
    const _Float16* Bbase = Bw + (long)nrow * Kp + kbB;

    v8f acc0 = {};
    v8f acc1 = {};
    for (int k = 0; k < Kp; k += 64) {
        // K slice [k, k+32)
        v8h a0 = *(const v8h*)(Abase + k);
        v8h a1 = *(const v8h*)(Abase + k + 16);
        v8h b0l = *(const v8h*)(Bbase + k);
        v8h b0h = *(const v8h*)(Bbase + k + 16);
        v16h b0 = __builtin_shufflevector(b0l, b0h, 0,1,2,3,4,5,6,7,8,9,10,11,12,13,14,15);
        v16h af0;
#pragma unroll
        for (int i = 0; i < 8; ++i) { af0[i] = a0[i]; af0[8 + i] = a1[i]; }
        // K slice [k+32, k+64)
        v8h a2 = *(const v8h*)(Abase + k + 32);
        v8h a3 = *(const v8h*)(Abase + k + 48);
        v8h b1l = *(const v8h*)(Bbase + k + 32);
        v8h b1h = *(const v8h*)(Bbase + k + 48);
        v16h b1 = __builtin_shufflevector(b1l, b1h, 0,1,2,3,4,5,6,7,8,9,10,11,12,13,14,15);
        v16h af1;
#pragma unroll
        for (int i = 0; i < 8; ++i) { af1[i] = a2[i]; af1[8 + i] = a3[i]; }

        acc0 = __builtin_amdgcn_wmma_f32_16x16x32_f16(
            false, af0, false, b0, (short)0, acc0, false, false);
        acc1 = __builtin_amdgcn_wmma_f32_16x16x32_f16(
            false, af1, false, b1, (short)0, acc1, false, false);
    }

    int n   = Ntile * 16 + (lane & 15);
    int mof = (lane >> 4) * 8;
    float gn = gam[n], bn_ = bet[n], mn = mu[n], bi = bias[n];
    float rs = rsqrtf(var[n] + BN_EPS);
#pragma unroll
    for (int r = 0; r < 8; ++r) {
        float v = acc0[r] + acc1[r] + bi;
        v = gn * (v - mn) * rs + bn_;
        v = v > 0.f ? v : 0.f;
        out[(long)(Mtile * 16 + mof + r) * DDIM + n] = (OUT)v;
    }
}

// ---------------- scalar head: logits, age loss, softmax head, BCE, final loss ----------------
__global__ void k_head(const float* __restrict__ feat, const float* __restrict__ age,
                       const float* __restrict__ labels,
                       const float* __restrict__ Wp, const float* __restrict__ bp,
                       const float* __restrict__ Wa, const float* __restrict__ ba,
                       const float* __restrict__ A1, const float* __restrict__ c1,
                       const float* __restrict__ A2, const float* __restrict__ c2,
                       float* __restrict__ out) {
    int b = threadIdx.x;                  // 0..127
    __shared__ float s_age[B], s_bce[B], s_msk[B];

    const float* f = feat + b * DDIM;
    float lo = 0.f, la = 0.f;
#pragma unroll 8
    for (int d = 0; d < DDIM; ++d) { lo += f[d] * Wp[d]; la += f[d] * Wa[d]; }
    lo += bp[0]; la += ba[0];

    float age_norm = (age[b] - 40.f) / 30.f;
    float age_pred = 1.f / (1.f + expf(-la));
    s_age[b] = fabsf(age_pred - age_norm);

    float pt0 = c2[0], pt1 = c2[1];
#pragma unroll
    for (int i = 0; i < 8; ++i) {
        float r = age_norm * A1[i] + c1[i];
        r = r > 0.f ? r : 0.f;
        pt0 += r * A2[i];
        pt1 += r * A2[8 + i];
    }
    float mx = fmaxf(pt0, pt1);
    float e0 = expf(pt0 - mx), e1 = expf(pt1 - mx);
    float pp1 = e1 / (e0 + e1);

    float p = 1.f / (1.f + expf(-lo));
    p = fminf(fmaxf(p, 1e-7f), 1.f - 1e-7f);
    float up = p * pp1;    // (1-p)*row0[1] + p*pos_probs[1], row0[1]==0

    float bce = -(p * logf(up) + (1.f - p) * log1pf(-up));
    float msk = (labels[b] == 1.f) ? 1.f : 0.f;
    s_bce[b] = bce * msk;
    s_msk[b] = msk;

    float lab = labels[b];
    float fp = lab * p + (1.f - lab) * up;   // (1-neg_mask)*p + neg_mask*up
    out[b]       = logf(fp / (1.f - fp + 1e-7f));
    out[B + b]   = lo;

    __syncthreads();
    for (int s = B / 2; s > 0; s >>= 1) {
        if (b < s) {
            s_age[b] += s_age[b + s];
            s_bce[b] += s_bce[b + s];
            s_msk[b] += s_msk[b + s];
        }
        __syncthreads();
    }
    if (b == 0) out[2 * B] = s_bce[0] / s_msk[0] + 0.5f * (s_age[0] / (float)B);
}

extern "C" void kernel_launch(void* const* d_in, const int* in_sizes, int n_in,
                              void* d_out, int out_size, void* d_ws, size_t ws_size,
                              hipStream_t stream) {
    (void)in_sizes; (void)n_in; (void)out_size; (void)ws_size;
    const float* snp     = (const float*)d_in[0];
    const int*   snp_ids = (const int*)d_in[1];
    const int*   seg     = (const int*)d_in[2];
    const float* age     = (const float*)d_in[3];
    const float* labels  = (const float*)d_in[4];
    const float* filters = (const float*)d_in[5];
    const float* W1 = (const float*)d_in[6];  const float* b1 = (const float*)d_in[7];
    const float* g1 = (const float*)d_in[8];  const float* be1 = (const float*)d_in[9];
    const float* m1 = (const float*)d_in[10]; const float* v1 = (const float*)d_in[11];
    const float* W2 = (const float*)d_in[12]; const float* b2 = (const float*)d_in[13];
    const float* g2 = (const float*)d_in[14]; const float* be2 = (const float*)d_in[15];
    const float* m2 = (const float*)d_in[16]; const float* v2 = (const float*)d_in[17];
    const float* Wp = (const float*)d_in[18]; const float* bp = (const float*)d_in[19];
    const float* Wa = (const float*)d_in[20]; const float* ba = (const float*)d_in[21];
    const float* A1 = (const float*)d_in[22]; const float* c1 = (const float*)d_in[23];
    const float* A2 = (const float*)d_in[24]; const float* c2 = (const float*)d_in[25];

    char* ws = (char*)d_ws;
    size_t off = 0;
    auto alloc = [&](size_t bytes) -> void* {
        void* p = ws + off;
        off = (off + bytes + 255) & ~(size_t)255;
        return p;
    };
    float*     fm   = (float*)alloc((size_t)N_SNPS * 4);
    int*       gs   = (int*)alloc((size_t)(N_GENES + 1) * 4);
    _Float16*  Ah   = (_Float16*)alloc((size_t)B * KP * 2);
    _Float16*  W1h  = (_Float16*)alloc((size_t)DDIM * KP * 2);
    _Float16*  W2h  = (_Float16*)alloc((size_t)DDIM * DDIM * 2);
    _Float16*  Hh   = (_Float16*)alloc((size_t)B * DDIM * 2);
    float*     feat = (float*)alloc((size_t)B * DDIM * 4);

    k_filt_mean<<<(N_SNPS + 255) / 256, 256, 0, stream>>>(filters, fm);
    k_gene_start<<<(N_GENES + 1 + 255) / 256, 256, 0, stream>>>(seg, gs);
    k_gather<<<(int)(((long)B * KP + 255) / 256), 256, 0, stream>>>(snp, snp_ids, gs, fm, Ah);
    k_cvt_f16<<<(int)(((long)DDIM * KP + 255) / 256), 256, 0, stream>>>(W1, W1h, DDIM, N_GENES, KP);
    k_cvt_f16<<<(DDIM * DDIM + 255) / 256, 256, 0, stream>>>(W2, W2h, DDIM, DDIM, DDIM);

    dim3 gtiles(B / 16, DDIM / 16);  // 8 x 4 tiles, one wave each
    k_gemm_bn_relu<_Float16><<<gtiles, 32, 0, stream>>>(Ah, W1h, b1, g1, be1, m1, v1, Hh, KP);
    k_gemm_bn_relu<float><<<gtiles, 32, 0, stream>>>(Hh, W2h, b2, g2, be2, m2, v2, feat, DDIM);

    k_head<<<1, B, 0, stream>>>(feat, age, labels, Wp, bp, Wa, ba, A1, c1, A2, c2, (float*)d_out);
}
